// PatternOperation_42580305773202
// MI455X (gfx1250) — matrix operand order repaired and checked by reading, not compiled, not run on hardware
//
#include <hip/hip_runtime.h>

typedef __attribute__((ext_vector_type(16))) _Float16 v16h;
typedef __attribute__((ext_vector_type(8)))  float    v8f;
typedef __attribute__((ext_vector_type(4)))  float    v4f;

#define TILE 28            // output tile edge
#define REG  32            // grid region edge  = TILE + 4 (window 3 + dilation 2)
#define MREG 30            // match region edge = TILE + 2
#define NPIX (REG * REG)   // 1024 region pixels
#define CPAD 24            // halves per pixel row in LDS (48B, 16B-aligned)
#define HGT  256
#define WID  256
#define NCH  10
#define HO   (HGT - 3 + 1) // 254

// ---------------------------------------------------------------------------
// Kernel 1: zero-fill the whole output with wide non-temporal stores.
// 9 of 10 channels are zeros; emitting them at b128 width is bandwidth-optimal.
// Channel target_color is overwritten by kernel 2 (same stream => ordered).
// ---------------------------------------------------------------------------
__global__ __launch_bounds__(256) void pattern_zero_kernel(v4f* __restrict__ out, int n4) {
    int i = blockIdx.x * 256 + threadIdx.x;
    if (i < n4) {
        v4f z = {};
        __builtin_nontemporal_store(z, &out[i]);
    }
}

// ---------------------------------------------------------------------------
// Kernel 2: fused match + dilate, WMMA channel contraction.
//   mismatch(y,x) = S + sum_o T[o, y+dy_o, x+dx_o],  S = sum(pattern)
//   T[pixel,o]    = sum_c g[c,pixel] * (1 - 2*pattern[c,o])   (exact in f16/f32)
// ---------------------------------------------------------------------------
__global__ __launch_bounds__(256) void pattern_match_kernel(
    const float* __restrict__ grid, const float* __restrict__ pattern,
    const int* __restrict__ tcolor, float* __restrict__ out) {

    __shared__ __attribute__((aligned(16))) _Float16 gLDS[NPIX * CPAD]; // 48 KB
    __shared__ __attribute__((aligned(8)))  _Float16 tLDS[9 * NPIX];    // 18 KB
    __shared__ float mLDS[MREG * MREG];                                 // 3.6 KB

    const int tid  = threadIdx.x;        // 0..255 (8 wave32s)
    const int lane = tid & 31;
    const int wave = tid >> 5;
    const int b    = blockIdx.z;
    const int y0   = blockIdx.y * TILE;
    const int x0   = blockIdx.x * TILE;
    const int ry0  = y0 - 2, rx0 = x0 - 2;   // region origin (match+dilate halo)

    // ---- Stage 1: grid region (REG x REG x 10ch) -> LDS f16, [pixel][channel]
    for (int c = 0; c < NCH; ++c) {
        const size_t base = (((size_t)b * NCH + c) << 16);
        for (int p = tid; p < NPIX; p += 256) {
            int ry = p >> 5, rx = p & 31;
            int gy = ry0 + ry, gx = rx0 + rx;
            float v = 0.f;
            if ((unsigned)gy < (unsigned)HGT && (unsigned)gx < (unsigned)WID) {
                v = grid[base + ((size_t)gy << 8) + gx];
                if (c + 1 < NCH && p == tid)  // stream one channel ahead
                    __builtin_prefetch(&grid[base + 0x10000u + ((size_t)gy << 8) + gx], 0, 1);
            }
            gLDS[p * CPAD + c] = (_Float16)v;
        }
    }
    // zero channels 10..15 so the A-fragment b128 load is branch-free for lanes>=16
    for (int p = tid; p < NPIX; p += 256) {
        unsigned int* gz = (unsigned int*)&gLDS[p * CPAD + 10];
        gz[0] = 0u; gz[1] = 0u; gz[2] = 0u;
    }

    // ---- Stage 2: constants from pattern (binary, exact)
    float S = 0.f;
    for (int i = 0; i < 90; ++i) S += pattern[i];

    // B fragment (32x16 f16): lane n (mod 16) = column o; lanes 0..15 hold K=0..7,
    // lanes 16..31 hold K=8..15 in elements 0..7 (K_real = 10); elements 8..15 zero.
    v16h bfrag = {};
    if ((lane & 15) < 9) {
        const int dy = (lane & 15) / 3, dx = (lane & 15) % 3;
#pragma unroll
        for (int c = 0; c < NCH; ++c)
            if ((c >> 3) == (lane >> 4))
                bfrag[c & 7] = (_Float16)(1.f - 2.f * pattern[c * 9 + dy * 3 + dx]);
    }
    __syncthreads();

    // ---- Stage 3: WMMA — T[pixel][o] for all 1024 region pixels (64 groups of 16)
    for (int g = wave; g < NPIX / 16; g += 8) {
        const int pb   = g * 16;
        const int prow = pb + (lane & 15);
        // A fragment (16x32 f16): lane L (0..15) row M=L holds K=0..7 in V0-3,
        // lane L+16 holds K=8..15 in V0-3 (c8,c9 real, c10..15 zero-padded in LDS);
        // V4-7 (K=16..31) = 0.  One aligned b128 per lane, address by lane half.
        const unsigned int* gp =
            (const unsigned int*)&gLDS[prow * CPAD + ((lane < 16) ? 0 : 8)];
        union { v16h h; unsigned int u[8]; } af;
        af.u[0] = gp[0]; af.u[1] = gp[1]; af.u[2] = gp[2]; af.u[3] = gp[3];
        af.u[4] = 0u;    af.u[5] = 0u;    af.u[6] = 0u;    af.u[7] = 0u;
        v8f d = {};
        d = __builtin_amdgcn_wmma_f32_16x16x32_f16(false, af.h, false, bfrag,
                                                   (short)0, d, false, false);
        // D layout: lane = column o (mod 16), VGPR r = row pb + r (+8 for lanes>=16)
        const int o = lane & 15;
        if (o < 9) {
            const int rbase = pb + ((lane < 16) ? 0 : 8);
            unsigned int* tp = (unsigned int*)&tLDS[o * NPIX + rbase];
#pragma unroll
            for (int r = 0; r < 8; r += 2) {
                unsigned short lo = __builtin_bit_cast(unsigned short, (_Float16)d[r]);
                unsigned short hi = __builtin_bit_cast(unsigned short, (_Float16)d[r + 1]);
                tp[r >> 1] = ((unsigned int)hi << 16) | (unsigned int)lo;
            }
        }
    }
    __syncthreads();

    // ---- Stage 4: match map on MREG x MREG (mismatch count == 0, valid anchors only)
    for (int q = tid; q < MREG * 32; q += 256) {       // 32-wide rows: shift/mask only
        const int ry = q >> 5, rx = q & 31;
        if (rx < MREG) {
            const int my = ry0 + ry, mx = rx0 + rx;
            float s = S;
#pragma unroll
            for (int o = 0; o < 9; ++o) {
                const int dy = o / 3, dx = o % 3;
                s += (float)tLDS[o * NPIX + (ry + dy) * REG + (rx + dx)];
            }
            const bool valid = ((unsigned)my < (unsigned)HO) && ((unsigned)mx < (unsigned)HO);
            mLDS[ry * MREG + rx] = (valid && s > -0.5f && s < 0.5f) ? 1.f : 0.f;
        }
    }
    __syncthreads();

    // ---- Stage 5: 3x3 max-dilation, store channel target_color (non-temporal)
    const int tc = tcolor[0];
    for (int q = tid; q < TILE * 32; q += 256) {       // 32-wide rows: shift/mask only
        const int oy = q >> 5, ox = q & 31;
        const int y = y0 + oy, x = x0 + ox;
        if (ox < TILE && y < HGT && x < WID) {
            float cov = 0.f;
#pragma unroll
            for (int dy = 0; dy < 3; ++dy)
#pragma unroll
                for (int dx = 0; dx < 3; ++dx)
                    cov = fmaxf(cov, mLDS[(oy + dy) * MREG + (ox + dx)]);
            __builtin_nontemporal_store(
                cov, &out[(((size_t)b * NCH + tc) << 16) + ((size_t)y << 8) + x]);
        }
    }
}

// ---------------------------------------------------------------------------
extern "C" void kernel_launch(void* const* d_in, const int* in_sizes, int n_in,
                              void* d_out, int out_size, void* d_ws, size_t ws_size,
                              hipStream_t stream) {
    const float* grid    = (const float*)d_in[0];   // (64,10,256,256) f32
    const float* pattern = (const float*)d_in[1];   // (1,10,3,3)      f32
    const int*   tcolor  = (const int*)d_in[2];     // scalar int
    float*       out     = (float*)d_out;           // (64,10,256,256) f32

    // 1) zero-fill full output with wide NT stores
    const int n4 = out_size / 4;                    // out_size divisible by 4
    pattern_zero_kernel<<<dim3((n4 + 255) / 256), dim3(256), 0, stream>>>((v4f*)out, n4);

    // 2) fused match + dilate, one block per (28x28 tile, batch)
    dim3 gridDim((WID + TILE - 1) / TILE, (HGT + TILE - 1) / TILE, 64); // (10,10,64)
    pattern_match_kernel<<<gridDim, dim3(256), 0, stream>>>(grid, pattern, tcolor, out);
}
